// ContrastiveLossELI5_30116310679581
// MI455X (gfx1250) — compile-verified, matrix operand order checked by reading, not run on hardware
//
#include <hip/hip_runtime.h>
#include <hip/hip_bf16.h>

#define N_BATCH 4096
#define DIM     512
#define TWO_N   8192
#define INV_T   2.0f   // 1 / TEMPERATURE, TEMPERATURE = 0.5

typedef __attribute__((ext_vector_type(16))) _Float16 v16h;
typedef __attribute__((ext_vector_type(8)))  _Float16 v8h;
typedef __attribute__((ext_vector_type(8)))  float    v8f;

// ---------------------------------------------------------------------------
// Fragment loader: 16x32 f16 tile slice for one wave, per CDNA5 ISA layout:
//   lane L (L<16) holds row/col (base + L), K = k0+0..7 (v[0..7]) and
//   K = k0+16..23 (v[8..15]); lane L+16 holds same row, K = k0+8..15 and
//   k0+24..31. Used identically for A and B (sim = Z * Z^T), so any hardware
//   position<->k permutation cancels between the two operands.
// ---------------------------------------------------------------------------
__device__ __forceinline__ v16h load_frag_g(const _Float16* __restrict__ z,
                                            int baseRow, int k0, int lane) {
    const _Float16* p = z + (size_t)(baseRow + (lane & 15)) * DIM
                          + k0 + ((lane >> 4) << 3);
    v8h lo = *(const v8h*)p;
    v8h hi = *(const v8h*)(p + 16);
    return __builtin_shufflevector(lo, hi, 0,1,2,3,4,5,6,7,8,9,10,11,12,13,14,15);
}

__device__ __forceinline__ v16h load_frag_s(const _Float16* As, int k0, int lane) {
    const _Float16* p = As + (lane & 15) * DIM + k0 + ((lane >> 4) << 3);
    v8h lo = *(const v8h*)p;
    v8h hi = *(const v8h*)(p + 16);
    return __builtin_shufflevector(lo, hi, 0,1,2,3,4,5,6,7,8,9,10,11,12,13,14,15);
}

// ---------------------------------------------------------------------------
// Kernel 1: L2-normalize rows of [emb_i; emb_j] -> f16 z[8192][512].
// One wave per row. Also zero-initializes denom[] / possim[] accumulators.
// ---------------------------------------------------------------------------
__global__ void __launch_bounds__(256)
norm_kernel(const float* __restrict__ emb_i, const float* __restrict__ emb_j,
            _Float16* __restrict__ z, float* __restrict__ denom,
            float* __restrict__ possim) {
    const int row  = blockIdx.x * (blockDim.x >> 5) + (threadIdx.x >> 5);
    const int lane = threadIdx.x & 31;
    const float* src = (row < N_BATCH) ? (emb_i + (size_t)row * DIM)
                                       : (emb_j + (size_t)(row - N_BATCH) * DIM);
    float vals[DIM / 32];
    float ss = 0.f;
#pragma unroll
    for (int i = 0; i < DIM / 32; ++i) {
        float v = src[lane + i * 32];
        vals[i] = v;
        ss += v * v;
    }
#pragma unroll
    for (int m = 16; m >= 1; m >>= 1) ss += __shfl_xor(ss, m, 32);
    const float inv = 1.0f / fmaxf(sqrtf(ss), 1e-12f);
    _Float16* dst = z + (size_t)row * DIM;
#pragma unroll
    for (int i = 0; i < DIM / 32; ++i)
        dst[lane + i * 32] = (_Float16)(vals[i] * inv);
    if (lane == 0) { denom[row] = 0.f; possim[row] = 0.f; }
}

// ---------------------------------------------------------------------------
// Kernel 2: tiled Z*Z^T with v_wmma_f32_16x16x32_f16, fused exp/row-sum.
// Block = 8 waves. Block b: column segment = b&3 (128 16-wide tiles),
// row strips = (b>>2)*8 + waveId (16 rows each). A strips staged in LDS;
// B tiles streamed from global, waves barrier-synced per column group so
// the WGP cache serves cross-wave B reuse.
// ---------------------------------------------------------------------------
__global__ void __launch_bounds__(256)
simloss_kernel(const _Float16* __restrict__ z, float* __restrict__ denom,
               float* __restrict__ possim) {
    __shared__ _Float16 Abuf[8 * 16 * DIM];   // 128 KB

    const int lane = threadIdx.x & 31;
    const int wv   = threadIdx.x >> 5;
    const int seg  = blockIdx.x & 3;
    const int rt   = (blockIdx.x >> 2) * 8 + wv;   // row-strip index 0..511
    const int R    = rt * 16;
    const int ct0  = seg * 128;
    const int ct1  = ct0 + 128;

    // Stage this wave's 16x512 A strip into LDS (16 KB per wave).
    {
        _Float16*       As = &Abuf[wv * 16 * DIM];
        const _Float16* gz = z + (size_t)R * DIM;
        for (int i = lane; i < (16 * DIM) / 8; i += 32)
            ((v8h*)As)[i] = ((const v8h*)gz)[i];
    }
    __syncthreads();
    const _Float16* As = &Abuf[wv * 16 * DIM];

    float rsum[8], pacc[8];
#pragma unroll
    for (int r = 0; r < 8; ++r) { rsum[r] = 0.f; pacc[r] = 0.f; }

    const int rowOff = ((lane >> 4) << 3);     // +0 or +8
    const int colSub = (lane & 15);

    for (int ct = ct0; ct < ct1; ct += 4) {
        v8f c0 = {}; v8f c1 = {}; v8f c2 = {}; v8f c3 = {};
#pragma unroll 2
        for (int kk = 0; kk < DIM; kk += 32) {
            v16h a  = load_frag_s(As, kk, lane);
            v16h b0 = load_frag_g(z, (ct + 0) * 16, kk, lane);
            v16h b1 = load_frag_g(z, (ct + 1) * 16, kk, lane);
            v16h b2 = load_frag_g(z, (ct + 2) * 16, kk, lane);
            v16h b3 = load_frag_g(z, (ct + 3) * 16, kk, lane);
            c0 = __builtin_amdgcn_wmma_f32_16x16x32_f16(false, a, false, b0,
                                                        (short)0, c0, false, false);
            c1 = __builtin_amdgcn_wmma_f32_16x16x32_f16(false, a, false, b1,
                                                        (short)0, c1, false, false);
            c2 = __builtin_amdgcn_wmma_f32_16x16x32_f16(false, a, false, b2,
                                                        (short)0, c2, false, false);
            c3 = __builtin_amdgcn_wmma_f32_16x16x32_f16(false, a, false, b3,
                                                        (short)0, c3, false, false);
        }
        // Fused epilogue: exp(sim/T) row-sum (excluding diagonal) + positive-pair capture.
#pragma unroll
        for (int t = 0; t < 4; ++t) {
            v8f c = (t == 0) ? c0 : (t == 1) ? c1 : (t == 2) ? c2 : c3;
            const int col = (ct + t) * 16 + colSub;
#pragma unroll
            for (int r = 0; r < 8; ++r) {
                const int   row = R + r + rowOff;
                const float val = c[r];
                const float e   = __expf(val * INV_T);
                rsum[r] += (row == col) ? 0.f : e;
                const int pcol = (row < N_BATCH) ? row + N_BATCH : row - N_BATCH;
                pacc[r] += (col == pcol) ? val : 0.f;
            }
        }
        __syncthreads();   // keep the 8 waves in lockstep for WGP$ B-tile reuse
    }

    // Reduce partial row sums across each 16-lane group (columns).
#pragma unroll
    for (int r = 0; r < 8; ++r) {
#pragma unroll
        for (int m = 1; m <= 8; m <<= 1) {
            rsum[r] += __shfl_xor(rsum[r], m, 32);
            pacc[r] += __shfl_xor(pacc[r], m, 32);
        }
    }
    if ((lane & 15) == 0) {
#pragma unroll
        for (int r = 0; r < 8; ++r) {
            const int row = R + r + rowOff;
            atomicAdd(&denom[row],  rsum[r]);
            atomicAdd(&possim[row], pacc[r]);
        }
    }
}

// ---------------------------------------------------------------------------
// Kernel 3: loss = ( sum_r log(denom_r) - (1/T) * sum_r pos_r ) / 2N
// ---------------------------------------------------------------------------
__global__ void __launch_bounds__(256)
finalize_kernel(const float* __restrict__ denom, const float* __restrict__ possim,
                float* __restrict__ out) {
    __shared__ float sbuf[256];
    float acc = 0.f;
    for (int r = threadIdx.x; r < TWO_N; r += 256)
        acc += __logf(denom[r]) - possim[r] * INV_T;
    sbuf[threadIdx.x] = acc;
    __syncthreads();
    for (int s = 128; s > 0; s >>= 1) {
        if (threadIdx.x < s) sbuf[threadIdx.x] += sbuf[threadIdx.x + s];
        __syncthreads();
    }
    if (threadIdx.x == 0) out[0] = sbuf[0] / (float)TWO_N;
}

// ---------------------------------------------------------------------------
extern "C" void kernel_launch(void* const* d_in, const int* in_sizes, int n_in,
                              void* d_out, int out_size, void* d_ws, size_t ws_size,
                              hipStream_t stream) {
    const float* emb_i = (const float*)d_in[0];
    const float* emb_j = (const float*)d_in[1];
    float*       out   = (float*)d_out;

    _Float16* z      = (_Float16*)d_ws;                         // 8 MB
    float*    denom  = (float*)((char*)d_ws + (size_t)TWO_N * DIM * sizeof(_Float16));
    float*    possim = denom + TWO_N;                           // +64 KB total

    // 1) normalize + convert to f16, zero accumulators (8192 waves)
    norm_kernel<<<dim3(TWO_N / 8), dim3(256), 0, stream>>>(emb_i, emb_j, z,
                                                           denom, possim);
    // 2) WMMA GEMM + fused softmax-denominator / positive capture
    //    256 blocks = 64 strip-groups x 4 column segments
    simloss_kernel<<<dim3(256), dim3(256), 0, stream>>>(z, denom, possim);
    // 3) scalar loss
    finalize_kernel<<<dim3(1), dim3(256), 0, stream>>>(denom, possim, out);
}
